// PerceiverAttentionLayer_3968549782269
// MI455X (gfx1250) — compile-verified, matrix operand order checked by reading, not run on hardware
//
#include <hip/hip_runtime.h>

// Perceiver cross-attention layer for MI455X (gfx1250, wave32, WMMA).
// All GEMMs use v_wmma_f32_16x16x32_bf16 (bf16 in, f32 accumulate).
// A-tiles / K-chunks are moved with GLOBAL_LOAD_ASYNC_TO_LDS_B128 and
// double-buffered in LDS (ASYNCcnt), overlapping HBM fetch with WMMA.

#define Bsz 16
#define Fdim 4096
#define Qn 64
#define DIM 512
#define Hn 8
#define DHd 64

typedef __attribute__((ext_vector_type(16))) __bf16 v16bf;
typedef __attribute__((ext_vector_type(8)))  float  v8f;
typedef __attribute__((ext_vector_type(4)))  float  v4f;

__device__ inline v8f wmma_bf16(v16bf a, v16bf b, v8f c) {
  // (neg_a, A, neg_b, B, c_mod, C, reuse_a, reuse_b)
  return __builtin_amdgcn_wmma_f32_16x16x32_bf16(false, a, false, b, (short)0, c, false, false);
}

// Build a 16x32 bf16 A-fragment (or B-fragment from a transposed tile).
// CDNA5 layout: lane<16 holds k in {0..7}U{16..23}; lane>=16 holds
// k in {8..15}U{24..31}. Two contiguous 16-byte chunks per lane.
__device__ inline v16bf frag_ld(const __bf16* rowbase, int kbase) {
  v16bf a;
  v4f* ap = reinterpret_cast<v4f*>(&a);
  ap[0] = *reinterpret_cast<const v4f*>(rowbase + kbase);
  ap[1] = *reinterpret_cast<const v4f*>(rowbase + kbase + 16);
  return a;
}

// Async 16-byte global->LDS copy (per lane), tracked by ASYNCcnt.
__device__ inline void async_cp16(const __bf16* lds_dst, const __bf16* gsrc) {
  unsigned lds = (unsigned)(unsigned long long)lds_dst;  // low 32b = LDS offset
  unsigned long long g = (unsigned long long)gsrc;
  asm volatile("global_load_async_to_lds_b128 %0, %1, off"
               :: "v"(lds), "v"(g) : "memory");
}
__device__ inline void wait_async0() {
  asm volatile("s_wait_asynccnt 0x0" ::: "memory");
}

// ---------------------------------------------------------------- cvt f32->bf16
__global__ __launch_bounds__(256) void cvt_kernel(const float* __restrict__ in,
                                                  __bf16* __restrict__ out, int n) {
  int i = blockIdx.x * blockDim.x + threadIdx.x;
  int stride = gridDim.x * blockDim.x;
  for (; i < n; i += stride) out[i] = (__bf16)in[i];
}

// ---------------------------------------------------------------- LayerNorm
// One wave (32 lanes) per row of 512; block = 8 rows. Output bf16.
__global__ __launch_bounds__(256) void ln_kernel(const float* __restrict__ X,
                                                 const float* __restrict__ g,
                                                 const float* __restrict__ bta,
                                                 __bf16* __restrict__ Y, int rows) {
  int w = threadIdx.x >> 5, lane = threadIdx.x & 31;
  int row = blockIdx.x * 8 + w;
  if (row >= rows) return;
  const float* xr = X + (size_t)row * DIM;
  float4 v[4];
  float s = 0.f, ss = 0.f;
#pragma unroll
  for (int k = 0; k < 4; ++k) {
    v[k] = reinterpret_cast<const float4*>(xr)[k * 32 + lane];
    s  += v[k].x + v[k].y + v[k].z + v[k].w;
    ss += v[k].x*v[k].x + v[k].y*v[k].y + v[k].z*v[k].z + v[k].w*v[k].w;
  }
#pragma unroll
  for (int off = 1; off < 32; off <<= 1) {
    s  += __shfl_xor(s,  off, 32);
    ss += __shfl_xor(ss, off, 32);
  }
  float mean = s * (1.0f / DIM);
  float var  = ss * (1.0f / DIM) - mean * mean;
  float rs   = rsqrtf(var + 1e-5f);
  __bf16* yr = Y + (size_t)row * DIM;
#pragma unroll
  for (int k = 0; k < 4; ++k) {
    int c = (k * 32 + lane) * 4;
    float4 gv = reinterpret_cast<const float4*>(g)[k * 32 + lane];
    float4 bv = reinterpret_cast<const float4*>(bta)[k * 32 + lane];
    yr[c + 0] = (__bf16)((v[k].x - mean) * rs * gv.x + bv.x);
    yr[c + 1] = (__bf16)((v[k].y - mean) * rs * gv.y + bv.y);
    yr[c + 2] = (__bf16)((v[k].z - mean) * rs * gv.z + bv.z);
    yr[c + 3] = (__bf16)((v[k].w - mean) * rs * gv.w + bv.w);
  }
}

// ---------------------------------------------------------------- WMMA GEMM
// C[M,N] = A[M,K] @ B[K,N], bf16 inputs, f32 accumulate.
// Block tile 256x64, 8 waves, each wave owns a 32x64 strip (2 A-frags reuse
// every B-frag). K step 32, LDS double-buffered; A tile moved with async
// global->LDS B128 copies, B tile prefetched to VGPRs during compute and
// scattered transposed afterwards. DUAL: second B / second C share the A tile.
#define MT 256
#define LDA 40  // LDS row stride (bf16 elems), 80B = 16B-aligned, conflict-padded
template <int DUAL, int RELU, int OUTF32>
__global__ __launch_bounds__(256) void gemm_kernel(
    const __bf16* __restrict__ A, const __bf16* __restrict__ B0,
    const __bf16* __restrict__ B1, void* __restrict__ C0, void* __restrict__ C1,
    int M, int N, int K, float scale) {
  __shared__ __bf16 As[2][MT * LDA];
  __shared__ __bf16 Bs0[2][64 * LDA];
  __shared__ __bf16 Bs1[DUAL ? 2 : 1][64 * LDA];

  const int tid = threadIdx.x;
  const int w = tid >> 5, lane = tid & 31;
  const int nlane = lane & 15;
  const int hi8 = (lane >= 16) ? 8 : 0;  // also the k-base of A/B fragments
  const int m0 = blockIdx.y * MT;
  const int n0 = blockIdx.x * 64;

  v8f acc0[2][4] = {};
  v8f acc1[2][4] = {};

  auto loadA_async = [&](int buf, int kk) {
#pragma unroll
    for (int s2 = 0; s2 < 4; ++s2) {
      int row = (tid >> 2) + 64 * s2;
      int i = (tid & 3) * 8;
      async_cp16(&As[buf][row * LDA + i], A + (size_t)(m0 + row) * K + kk + i);
    }
  };
  auto loadB_g = [&](const __bf16* Bsrc, int kk) -> uint4 {
    int k = tid >> 3, i = (tid & 7) * 8;
    return *reinterpret_cast<const uint4*>(Bsrc + (size_t)(kk + k) * N + n0 + i);
  };
  auto scatB = [&](__bf16* dst, uint4 d) {  // transpose scatter: Bs[n][k]
    int k = tid >> 3, i = (tid & 7) * 8;
    const __bf16* dp = reinterpret_cast<const __bf16*>(&d);
#pragma unroll
    for (int t = 0; t < 8; ++t) dst[(i + t) * LDA + k] = dp[t];
  };

  // prologue: stage k-chunk 0
  loadA_async(0, 0);
  {
    uint4 d0 = loadB_g(B0, 0);
    scatB(Bs0[0], d0);
    if (DUAL) { uint4 d1 = loadB_g(B1, 0); scatB(Bs1[0], d1); }
  }
  wait_async0();
  __syncthreads();

  const int nk = K / 32;
  for (int it = 0; it < nk; ++it) {
    const int cur = it & 1, nxt = cur ^ 1;
    const bool more = (it + 1 < nk);
    uint4 d0 = {}, d1 = {};
    if (more) {  // start next chunk moving while we compute
      loadA_async(nxt, (it + 1) * 32);
      d0 = loadB_g(B0, (it + 1) * 32);
      if (DUAL) d1 = loadB_g(B1, (it + 1) * 32);
    }

    v16bf af0 = frag_ld(&As[cur][(w * 32 + nlane) * LDA], hi8);
    v16bf af1 = frag_ld(&As[cur][(w * 32 + 16 + nlane) * LDA], hi8);
#pragma unroll
    for (int j = 0; j < 4; ++j) {
      v16bf bf0 = frag_ld(&Bs0[cur][(16 * j + nlane) * LDA], hi8);
      acc0[0][j] = wmma_bf16(af0, bf0, acc0[0][j]);
      acc0[1][j] = wmma_bf16(af1, bf0, acc0[1][j]);
      if (DUAL) {
        v16bf bf1 = frag_ld(&Bs1[cur][(16 * j + nlane) * LDA], hi8);
        acc1[0][j] = wmma_bf16(af0, bf1, acc1[0][j]);
        acc1[1][j] = wmma_bf16(af1, bf1, acc1[1][j]);
      }
    }

    if (more) {
      scatB(Bs0[nxt], d0);
      if (DUAL) scatB(Bs1[nxt], d1);
    }
    wait_async0();
    __syncthreads();
  }

#pragma unroll
  for (int sub = 0; sub < 2; ++sub) {
#pragma unroll
    for (int j = 0; j < 4; ++j) {
#pragma unroll
      for (int r = 0; r < 8; ++r) {
        size_t m = (size_t)(m0 + w * 32 + sub * 16 + r + hi8);
        size_t n = (size_t)(n0 + 16 * j + nlane);
        float v0 = acc0[sub][j][r] * scale;
        if (RELU) v0 = fmaxf(v0, 0.f);
        if (OUTF32) reinterpret_cast<float*>(C0)[m * N + n] = v0;
        else        reinterpret_cast<__bf16*>(C0)[m * N + n] = (__bf16)v0;
        if (DUAL) {
          float v1 = acc1[sub][j][r] * scale;
          if (OUTF32) reinterpret_cast<float*>(C1)[m * N + n] = v1;
          else        reinterpret_cast<__bf16*>(C1)[m * N + n] = (__bf16)v1;
        }
      }
    }
  }
}

// ---------------------------------------------------------------- Attention
// One block per (b,h): Q tile 64x64 resident; stream K/V in chunks of 64 rows,
// double-buffered: K via async global->LDS, V prefetched to VGPRs during the
// current chunk's WMMAs and scattered transposed. Online (flash) softmax;
// mask==0 -> p=0; l==0 -> out=0 (nan_to_num).
#define LQK 72  // LDS row stride for 64-wide tiles (144B, 16B-aligned)
__global__ __launch_bounds__(128) void attn_kernel(
    const __bf16* __restrict__ Qb, const __bf16* __restrict__ Kb,
    const __bf16* __restrict__ Vb, const int* __restrict__ mask,
    float* __restrict__ Out) {
  __shared__ __bf16 Qs[64 * LQK];
  __shared__ __bf16 Ks[2][64 * LQK];
  __shared__ __bf16 Vt[2][64 * LQK];     // transposed: Vt[d][f]
  __shared__ __bf16 Ps[4 * 16 * LQK];    // per-wave P tiles

  const int tid = threadIdx.x;
  const int w = tid >> 5, lane = tid & 31;
  const int nlane = lane & 15;
  const int hi8 = (lane >= 16) ? 8 : 0;
  const int b = blockIdx.x >> 3, h = blockIdx.x & 7;
  const int f_ = tid >> 1, half = tid & 1;  // cooperative-load mapping

  auto loadK_async = [&](int buf, int fc) {
    size_t grow = (size_t)(b * Fdim + fc * 64 + f_) * DIM + h * DHd + half * 32;
#pragma unroll
    for (int i = 0; i < 4; ++i)
      async_cp16(&Ks[buf][f_ * LQK + half * 32 + i * 8], Kb + grow + i * 8);
  };
  auto loadV_g = [&](int fc, uint4* vv) {
    const uint4* src = reinterpret_cast<const uint4*>(
        Vb + (size_t)(b * Fdim + fc * 64 + f_) * DIM + h * DHd + half * 32);
#pragma unroll
    for (int i = 0; i < 4; ++i) vv[i] = src[i];
  };
  auto scatV = [&](int buf, const uint4* vv) {
    const __bf16* vp = reinterpret_cast<const __bf16*>(vv);
#pragma unroll
    for (int i = 0; i < 32; ++i) Vt[buf][(half * 32 + i) * LQK + f_] = vp[i];
  };

  {  // load Q tile (64 q rows x 64 d) + stage chunk 0
    const uint4* src = reinterpret_cast<const uint4*>(
        Qb + (size_t)(b * Qn + f_) * DIM + h * DHd + half * 32);
    uint4* dst = reinterpret_cast<uint4*>(Qs + f_ * LQK + half * 32);
#pragma unroll
    for (int i = 0; i < 4; ++i) dst[i] = src[i];
    loadK_async(0, 0);
    uint4 vv[4];
    loadV_g(0, vv);
    scatV(0, vv);
  }
  wait_async0();
  __syncthreads();

  v16bf qa0 = frag_ld(Qs + (w * 16 + nlane) * LQK + 0, hi8);
  v16bf qa1 = frag_ld(Qs + (w * 16 + nlane) * LQK + 32, hi8);

  v8f acc[4] = {};
  float mrun[8], lrun[8];
#pragma unroll
  for (int r = 0; r < 8; ++r) { mrun[r] = -1e30f; lrun[r] = 0.f; }

  __bf16* Pw = Ps + w * 16 * LQK;

  for (int fc = 0; fc < Fdim / 64; ++fc) {
    const int cur = fc & 1, nxt = cur ^ 1;
    const bool more = (fc + 1 < Fdim / 64);
    uint4 vv[4];
    if (more) {  // start next chunk moving
      loadK_async(nxt, fc + 1);
      loadV_g(fc + 1, vv);
    }

    // sim strip: 16(q) x 64(f) = 4 WMMA tiles, K=64 chained as 2x32
    v8f s[4];
    int mv[4];
#pragma unroll
    for (int j = 0; j < 4; ++j) {
      v16bf kb0 = frag_ld(&Ks[cur][(16 * j + nlane) * LQK + 0], hi8);
      v16bf kb1 = frag_ld(&Ks[cur][(16 * j + nlane) * LQK + 32], hi8);
      v8f z = {};
      s[j] = wmma_bf16(qa0, kb0, z);
      s[j] = wmma_bf16(qa1, kb1, s[j]);
      mv[j] = mask[b * Fdim + fc * 64 + 16 * j + nlane];
    }

    // online softmax per row (row r+hi8 lives in VGPR r across a 16-lane half)
#pragma unroll
    for (int r = 0; r < 8; ++r) {
      float sj[4];
      float cm = -1e30f;
#pragma unroll
      for (int j = 0; j < 4; ++j) {
        sj[j] = mv[j] ? s[j][r] : -1e30f;
        cm = fmaxf(cm, sj[j]);
      }
#pragma unroll
      for (int off = 1; off < 16; off <<= 1) cm = fmaxf(cm, __shfl_xor(cm, off, 32));
      float mn = fmaxf(mrun[r], cm);
      float cr = __expf(mrun[r] - mn);
      mrun[r] = mn;
      float rs = 0.f;
#pragma unroll
      for (int j = 0; j < 4; ++j) {
        float p = mv[j] ? __expf(sj[j] - mn) : 0.f;
        rs += p;
        Pw[(r + hi8) * LQK + 16 * j + nlane] = (__bf16)p;
      }
#pragma unroll
      for (int off = 1; off < 16; off <<= 1) rs += __shfl_xor(rs, off, 32);
      lrun[r] = lrun[r] * cr + rs;
#pragma unroll
      for (int t = 0; t < 4; ++t) acc[t][r] = acc[t][r] * cr;
    }

    // O += P @ V   (A = P 16x64 from LDS, B = V^T layout from Vt)
    v16bf pa0 = frag_ld(Pw + nlane * LQK + 0, hi8);
    v16bf pa1 = frag_ld(Pw + nlane * LQK + 32, hi8);
#pragma unroll
    for (int t = 0; t < 4; ++t) {
      v16bf vb0 = frag_ld(&Vt[cur][(16 * t + nlane) * LQK + 0], hi8);
      v16bf vb1 = frag_ld(&Vt[cur][(16 * t + nlane) * LQK + 32], hi8);
      acc[t] = wmma_bf16(pa0, vb0, acc[t]);
      acc[t] = wmma_bf16(pa1, vb1, acc[t]);
    }

    if (more) scatV(nxt, vv);
    wait_async0();
    __syncthreads();
  }

#pragma unroll
  for (int t = 0; t < 4; ++t) {
#pragma unroll
    for (int r = 0; r < 8; ++r) {
      int q = w * 16 + r + hi8;
      float l = lrun[r];
      float val = (l > 0.f) ? acc[t][r] / l : 0.f;
      Out[(size_t)(b * Qn + q) * DIM + h * DHd + 16 * t + nlane] = val;
    }
  }
}

// ---------------------------------------------------------------- launcher
extern "C" void kernel_launch(void* const* d_in, const int* in_sizes, int n_in,
                              void* d_out, int out_size, void* d_ws, size_t ws_size,
                              hipStream_t stream) {
  (void)in_sizes; (void)n_in; (void)out_size; (void)ws_size;
  const float* features = (const float*)d_in[0];
  const float* latents  = (const float*)d_in[1];
  const int*   mask     = (const int*)d_in[2];
  const float* gf = (const float*)d_in[3];
  const float* bf = (const float*)d_in[4];
  const float* gl = (const float*)d_in[5];
  const float* bl = (const float*)d_in[6];
  const float* Wq = (const float*)d_in[7];
  const float* Wk = (const float*)d_in[8];
  const float* Wv = (const float*)d_in[9];
  const float* gff = (const float*)d_in[10];
  const float* bff = (const float*)d_in[11];
  const float* W1 = (const float*)d_in[12];
  const float* W2 = (const float*)d_in[13];

  char* ws = (char*)d_ws;
  size_t off = 0;
  auto take = [&](size_t bytes) { char* p = ws + off; off += (bytes + 255) & ~(size_t)255; return p; };

  __bf16* xln   = (__bf16*)take((size_t)Bsz * Fdim * DIM * 2);
  __bf16* Kbuf  = (__bf16*)take((size_t)Bsz * Fdim * DIM * 2);
  __bf16* Vbuf  = (__bf16*)take((size_t)Bsz * Fdim * DIM * 2);
  __bf16* latln = (__bf16*)take((size_t)Bsz * Qn * DIM * 2);
  __bf16* Qbuf  = (__bf16*)take((size_t)Bsz * Qn * DIM * 2);
  float*  attno = (float*) take((size_t)Bsz * Qn * DIM * 4);
  __bf16* lnff  = (__bf16*)take((size_t)Bsz * Qn * DIM * 2);
  __bf16* hbuf  = (__bf16*)take((size_t)Bsz * Qn * 1024 * 2);
  __bf16* WqB   = (__bf16*)take((size_t)DIM * DIM * 2);
  __bf16* WkB   = (__bf16*)take((size_t)DIM * DIM * 2);
  __bf16* WvB   = (__bf16*)take((size_t)DIM * DIM * 2);
  __bf16* W1B   = (__bf16*)take((size_t)DIM * 1024 * 2);
  __bf16* W2B   = (__bf16*)take((size_t)1024 * DIM * 2);

  // weights -> bf16
  cvt_kernel<<<512, 256, 0, stream>>>(Wq, WqB, DIM * DIM);
  cvt_kernel<<<512, 256, 0, stream>>>(Wk, WkB, DIM * DIM);
  cvt_kernel<<<512, 256, 0, stream>>>(Wv, WvB, DIM * DIM);
  cvt_kernel<<<512, 256, 0, stream>>>(W1, W1B, DIM * 1024);
  cvt_kernel<<<512, 256, 0, stream>>>(W2, W2B, 1024 * DIM);

  // LayerNorms -> bf16
  ln_kernel<<<(Bsz * Fdim) / 8, 256, 0, stream>>>(features, gf, bf, xln, Bsz * Fdim);
  ln_kernel<<<(Bsz * Qn) / 8, 256, 0, stream>>>(latents, gl, bl, latln, Bsz * Qn);

  // K,V = xln @ Wk, xln @ Wv  (single pass over features, dual-B)
  gemm_kernel<1, 0, 0><<<dim3(DIM / 64, (Bsz * Fdim) / MT), 256, 0, stream>>>(
      xln, WkB, WvB, Kbuf, Vbuf, Bsz * Fdim, DIM, DIM, 1.0f);

  // Q = latln @ Wq, scaled by DH^-0.5 folded into the store
  gemm_kernel<0, 0, 0><<<dim3(DIM / 64, (Bsz * Qn) / MT), 256, 0, stream>>>(
      latln, WqB, nullptr, Qbuf, nullptr, Bsz * Qn, DIM, DIM, 0.125f);

  // flash attention per (b,h)
  attn_kernel<<<Bsz * Hn, 128, 0, stream>>>(Qbuf, Kbuf, Vbuf, mask, attno);

  // FF: LN -> W1 + ReLU -> W2
  ln_kernel<<<(Bsz * Qn) / 8, 256, 0, stream>>>(attno, gff, bff, lnff, Bsz * Qn);
  gemm_kernel<0, 1, 0><<<dim3(1024 / 64, (Bsz * Qn) / MT), 256, 0, stream>>>(
      lnff, W1B, nullptr, hbuf, nullptr, Bsz * Qn, 1024, DIM, 1.0f);
  gemm_kernel<0, 0, 1><<<dim3(DIM / 64, (Bsz * Qn) / MT), 256, 0, stream>>>(
      hbuf, W2B, nullptr, d_out, nullptr, Bsz * Qn, DIM, 1024, 1.0f);
}
